// GPT_15049565405483
// MI455X (gfx1250) — matrix operand order repaired and checked by reading, not compiled, not run on hardware
//
#include <hip/hip_runtime.h>
#include <math.h>

typedef __bf16 bf16_t;
typedef __attribute__((ext_vector_type(16))) __bf16 v16bf;
typedef __attribute__((ext_vector_type(8)))  float  v8f;
typedef __attribute__((ext_vector_type(4)))  unsigned int v4u;
typedef __attribute__((ext_vector_type(8)))  int v8i32;
typedef __attribute__((ext_vector_type(4)))  int v4i32;

constexpr int BATCH = 2, SEQ = 2048, DM = 1024, NH = 16, HD = 64, NL = 8,
              FFD = 4096, VOC = 32000;
constexpr int BT = BATCH * SEQ;  // 4096

// ---------------------------------------------------------------------------
// CDNA5 feature detection (compile-time, with safe fallbacks)
// ---------------------------------------------------------------------------
#if defined(__has_builtin)
#if __has_builtin(__builtin_amdgcn_tensor_load_to_lds) && \
    __has_builtin(__builtin_amdgcn_s_wait_tensorcnt)
#define USE_TDM 1
#endif
#endif
#ifndef USE_TDM
#define USE_TDM 0
#endif

// ---------------------------------------------------------------------------
// WMMA helpers (CDNA5 16x16x32 bf16 -> f32)
// ---------------------------------------------------------------------------
__device__ inline v8f wmma_bf16(v16bf a, v16bf b, v8f c) {
  return __builtin_amdgcn_wmma_f32_16x16x32_bf16(
      /*neg_a=*/false, a, /*neg_b=*/false, b,
      /*c_mod=*/(short)0, c, /*reuse_a=*/false, /*reuse_b=*/false);
}

// A-matrix fragment (16x32 bf16): lane (0..15) = row M, lane>>4 picks khalf.
// element e -> k = 2*kg + e, kg = (lane>>4)*8 (contiguous, matches B).
__device__ inline v16bf ld_afrag(const bf16_t* rowp, int kg) {
  union { uint4 u[2]; v16bf v; } t;
  t.u[0] = *(const uint4*)(rowp + 2 * kg);
  t.u[1] = *(const uint4*)(rowp + 2 * kg + 8);
  return t.v;
}
// B-matrix fragment: 16 contiguous bf16 (element e -> k = khalf*16 + e).
__device__ inline v16bf ld_bfrag(const bf16_t* p) {
  union { uint4 u[2]; v16bf v; } t;
  t.u[0] = *(const uint4*)(p);
  t.u[1] = *(const uint4*)(p + 8);
  return t.v;
}

__device__ inline float gelu_exact(float x) {
  return 0.5f * x * (1.0f + erff(x * 0.70710678118654752f));
}
__device__ inline float hmax16(float v) {
#pragma unroll
  for (int off = 1; off < 16; off <<= 1) v = fmaxf(v, __shfl_xor(v, off, 32));
  return v;
}
__device__ inline float hsum16(float v) {
#pragma unroll
  for (int off = 1; off < 16; off <<= 1) v += __shfl_xor(v, off, 32);
  return v;
}

// ---------------------------------------------------------------------------
// Tensor Data Mover: 2D tile load (rows x rowbytes, byte-granular) -> LDS.
// D# bitfields per CDNA5 ISA 8.3/8.4: group0 = {count/type, lds_addr,
// global_addr[56:0], type=2}; group1 = {data_size=1B, tensor/tile dims,
// tensor_dim0_stride}. Groups 2/3 disabled (2D tensor). Issued by one wave,
// tracked by TENSORcnt. This toolchain exposes the 6-arg builtin form.
// ---------------------------------------------------------------------------
#if USE_TDM
__device__ inline unsigned lds_addr_of(const void* p) {
  return (unsigned)(unsigned long long)(
      (const __attribute__((address_space(3))) char*)p);
}

__device__ inline void tdm_load_2d(unsigned lds_off, const void* gptr,
                                   unsigned rows, unsigned rowbytes,
                                   unsigned long long stride_bytes) {
  const unsigned long long ga = (unsigned long long)gptr;
  v4u g0;
  g0[0] = 1u;                                   // count=1, user desc
  g0[1] = lds_off;                              // lds_addr (bytes)
  g0[2] = (unsigned)(ga & 0xffffffffu);         // global_addr[31:0]
  g0[3] = (unsigned)((ga >> 32) & 0x1ffffffu)   // global_addr[56:32]
          | 0x80000000u;                        // type=2 ("image")
  v8i32 g1;
  g1[0] = 0;                                    // wg_mask=0, data_size=1B
  g1[1] = (int)((rowbytes & 0xffffu) << 16);            // tensor_dim0 lo16
  g1[2] = (int)(((rowbytes >> 16) & 0xffffu) |          // tensor_dim0 hi16
                ((rows & 0xffffu) << 16));              // tensor_dim1 lo16
  g1[3] = (int)(((rows >> 16) & 0xffffu) |              // tensor_dim1 hi16
                ((rowbytes & 0xffffu) << 16));          // tile_dim0
  g1[4] = (int)(rows & 0xffffu);                        // tile_dim1 (dim2=0)
  g1[5] = (int)(stride_bytes & 0xffffffffu);            // dim0_stride lo32
  g1[6] = (int)((stride_bytes >> 32) & 0xffffu);        // dim0_stride hi16
  g1[7] = 0;
  const v4i32 z4 = {0, 0, 0, 0};
  const v8i32 z8 = {0, 0, 0, 0, 0, 0, 0, 0};
  __builtin_amdgcn_tensor_load_to_lds(g0, g1, z4, z4, z8, 0);
}
#endif

// ---------------------------------------------------------------------------
// Generic bf16 WMMA GEMM: out = A[M,K] @ Bt[N,K]^T (+bias)(+res)(GELU)(bf16/f32)
// B is pre-transposed to [N][K] by the weight-conversion kernels, so both A
// and B tiles are contiguous 128x64-byte 2D copies -> TDM-friendly.
// Block tile 128x128, BK=32, 256 threads = 8 waves (4x2), wave tile 32x64.
// Double-buffered: DMA/copy of tile k+1 overlaps WMMAs on tile k.
// ---------------------------------------------------------------------------
template <bool BIAS, bool RES, bool GELU_ACT, bool OUTBF16>
__global__ void __launch_bounds__(256)
gemm_bf16(const bf16_t* __restrict__ A, const bf16_t* __restrict__ Bt,
          const float* __restrict__ bias, const float* __restrict__ res,
          void* __restrict__ outv, int M, int N, int K) {
  __shared__ bf16_t As[2][128][32];  // [buf][m][k]
  __shared__ bf16_t Bs[2][128][32];  // [buf][n][k]
  const int tid = threadIdx.x;
  const int lane = tid & 31;
  const int wave = tid >> 5;
  const int wm = wave >> 1;  // 0..3 -> 32-row slab
  const int wn = wave & 1;   // 0..1 -> 64-col slab
  const int bm = blockIdx.y * 128;
  const int bn = blockIdx.x * 128;

  v8f acc[2][4];
#pragma unroll
  for (int i = 0; i < 2; ++i)
#pragma unroll
    for (int j = 0; j < 4; ++j)
#pragma unroll
      for (int r = 0; r < 8; ++r) acc[i][j][r] = 0.0f;

  const int fr = lane & 15;
  const int kg = (lane >> 4) * 8;
  const int kh = (lane >> 4) * 16;

  // tile staging (TDM: wave0 issues 2 DMA descriptors; else all-thread copy)
  auto stage = [&](int buf, int k0) {
#if USE_TDM
    if (tid < 32) {
      tdm_load_2d(lds_addr_of(&As[buf][0][0]),
                  A + (size_t)bm * K + k0, 128u, 64u,
                  (unsigned long long)K * 2u);
      tdm_load_2d(lds_addr_of(&Bs[buf][0][0]),
                  Bt + (size_t)bn * K + k0, 128u, 64u,
                  (unsigned long long)K * 2u);
    }
#else
    const int row = tid >> 1, c0 = (tid & 1) * 16;
    const uint4* ag = (const uint4*)(A + (size_t)(bm + row) * K + k0 + c0);
    *(uint4*)(&As[buf][row][c0]) = ag[0];
    *(uint4*)(&As[buf][row][c0 + 8]) = ag[1];
    const uint4* bg = (const uint4*)(Bt + (size_t)(bn + row) * K + k0 + c0);
    *(uint4*)(&Bs[buf][row][c0]) = bg[0];
    *(uint4*)(&Bs[buf][row][c0 + 8]) = bg[1];
    if (k0 + 32 < K) {  // gfx1250 global_prefetch_b8 of the following tile
      __builtin_prefetch(A + (size_t)(bm + row) * K + k0 + 32 + c0, 0, 0);
      __builtin_prefetch(Bt + (size_t)(bn + row) * K + k0 + 32 + c0, 0, 0);
    }
#endif
  };
  auto stage_wait = [&]() {
#if USE_TDM
    if (tid < 32) __builtin_amdgcn_s_wait_tensorcnt(0);
#endif
  };

  stage(0, 0);
  stage_wait();
  __syncthreads();

  int buf = 0;
  for (int k0 = 0; k0 < K; k0 += 32) {
    if (k0 + 32 < K) stage(buf ^ 1, k0 + 32);

    v16bf a0 = ld_afrag(&As[buf][wm * 32 + fr][0], kg);
    v16bf a1 = ld_afrag(&As[buf][wm * 32 + 16 + fr][0], kg);
#pragma unroll
    for (int fn = 0; fn < 4; ++fn) {
      v16bf bfr = ld_bfrag(&Bs[buf][wn * 64 + fn * 16 + fr][kh]);
      acc[0][fn] = wmma_bf16(a0, bfr, acc[0][fn]);
      acc[1][fn] = wmma_bf16(a1, bfr, acc[1][fn]);
    }

    stage_wait();
    __syncthreads();
    buf ^= 1;
  }

  // Epilogue. C layout: lane half -> rows 0-7 / 8-15, VGPR index = row.
  const int cr = (lane >> 4) * 8;
#pragma unroll
  for (int fm = 0; fm < 2; ++fm) {
#pragma unroll
    for (int fn = 0; fn < 4; ++fn) {
      const int gm0 = bm + wm * 32 + fm * 16 + cr;
      const int gn = bn + wn * 64 + fn * 16 + fr;
      float bv = 0.0f;
      if (BIAS) bv = bias[gn];
#pragma unroll
      for (int r = 0; r < 8; ++r) {
        float vv = acc[fm][fn][r] + bv;
        if (RES) vv += res[(size_t)(gm0 + r) * N + gn];
        if (GELU_ACT) vv = gelu_exact(vv);
        if (OUTBF16)
          ((bf16_t*)outv)[(size_t)(gm0 + r) * N + gn] = (bf16_t)vv;
        else
          ((float*)outv)[(size_t)(gm0 + r) * N + gn] = vv;
      }
    }
  }
}

// ---------------------------------------------------------------------------
// Flash attention: 1 workgroup (128 thr = 4 waves) per (b*h, 64-row q tile).
// Each wave owns 16 q rows; streams causal 64-row K/V blocks through LDS.
// S = Q K^T and O += P V both use WMMA; online softmax keeps row stats
// per-lane (C layout: rows live in VGPR index, cols across 16-lane groups).
// ---------------------------------------------------------------------------
__global__ void __launch_bounds__(128)
attention_kernel(const bf16_t* __restrict__ Q, const bf16_t* __restrict__ Kt,
                 const bf16_t* __restrict__ Vt, bf16_t* __restrict__ Ctx) {
  __shared__ bf16_t Ks[64][64];      // [krow][hd] (also Q staging)
  __shared__ bf16_t Vs[64][64];      // [hd][krow]
  __shared__ bf16_t Ps[4][16][64];   // per-wave P bounce (C->A layout)
  const int qt = blockIdx.x;
  const int bh = blockIdx.y;
  const int bb = bh >> 4;   // NH == 16
  const int hh = bh & 15;
  const int tid = threadIdx.x, lane = tid & 31, w = tid >> 5;
  const int fr = lane & 15;
  const int kg = (lane >> 4) * 8;
  const int kh = (lane >> 4) * 16;
  const int rowhalf = (lane >> 4) * 8;
  const bf16_t* Qh = Q + (size_t)bh * SEQ * HD;
  const bf16_t* Kh = Kt + (size_t)bh * SEQ * HD;
  const bf16_t* Vh = Vt + (size_t)bh * SEQ * HD;

  {  // stage Q tile (64x64) into Ks, pull A frags, then release Ks
    const int row = tid >> 1, c0 = (tid & 1) * 32;
    const uint4* src = (const uint4*)(Qh + (size_t)(qt * 64 + row) * HD + c0);
    uint4* dst = (uint4*)(&Ks[row][c0]);
#pragma unroll
    for (int i = 0; i < 4; ++i) dst[i] = src[i];
  }
  __syncthreads();
  v16bf qa0 = ld_afrag(&Ks[w * 16 + fr][0], kg);
  v16bf qa1 = ld_afrag(&Ks[w * 16 + fr][32], kg);
  __syncthreads();

  v8f o[4];
#pragma unroll
  for (int fn = 0; fn < 4; ++fn)
#pragma unroll
    for (int r = 0; r < 8; ++r) o[fn][r] = 0.0f;
  float m_[8], l_[8];
#pragma unroll
  for (int r = 0; r < 8; ++r) { m_[r] = -3.0e38f; l_[r] = 0.0f; }

  for (int kb = 0; kb <= qt; ++kb) {
    {  // load K block -> Ks[krow][hd]; V block transposed -> Vs[hd][krow]
      const int row = tid >> 1, c0 = (tid & 1) * 32;
      const uint4* ks = (const uint4*)(Kh + (size_t)(kb * 64 + row) * HD + c0);
      uint4* kd = (uint4*)(&Ks[row][c0]);
#pragma unroll
      for (int i = 0; i < 4; ++i) kd[i] = ks[i];
      const bf16_t* vsrc = Vh + (size_t)(kb * 64 + row) * HD + c0;
#pragma unroll
      for (int i = 0; i < 32; ++i) Vs[c0 + i][row] = vsrc[i];
    }
    __syncthreads();

    // S = Q K^T : B-matrix element [k=hd][n=krow] = K[krow][hd]
    v8f s[4];
#pragma unroll
    for (int fn = 0; fn < 4; ++fn) {
#pragma unroll
      for (int r = 0; r < 8; ++r) s[fn][r] = 0.0f;
      s[fn] = wmma_bf16(qa0, ld_bfrag(&Ks[fn * 16 + fr][kh]), s[fn]);
      s[fn] = wmma_bf16(qa1, ld_bfrag(&Ks[fn * 16 + fr][32 + kh]), s[fn]);
    }

    // scale 1/sqrt(64) + causal mask (arithmetic, EXEC stays all-ones)
    const int qrow0 = qt * 64 + w * 16 + rowhalf;
#pragma unroll
    for (int fn = 0; fn < 4; ++fn) {
      const int col = kb * 64 + fn * 16 + fr;
#pragma unroll
      for (int r = 0; r < 8; ++r) {
        float vv = s[fn][r] * 0.125f;
        if (col > qrow0 + r) vv = -3.0e38f;
        s[fn][r] = vv;
      }
    }

    // online softmax: row reductions = shfl_xor within 16-lane groups
    float nm[8], sc[8];
#pragma unroll
    for (int r = 0; r < 8; ++r) {
      float mx = fmaxf(fmaxf(s[0][r], s[1][r]), fmaxf(s[2][r], s[3][r]));
      mx = hmax16(mx);
      nm[r] = fmaxf(m_[r], mx);
      sc[r] = expf(m_[r] - nm[r]);
      m_[r] = nm[r];
    }
#pragma unroll
    for (int fn = 0; fn < 4; ++fn)
#pragma unroll
      for (int r = 0; r < 8; ++r) s[fn][r] = expf(s[fn][r] - nm[r]);
#pragma unroll
    for (int r = 0; r < 8; ++r) {
      float rs = s[0][r] + s[1][r] + s[2][r] + s[3][r];
      rs = hsum16(rs);
      l_[r] = l_[r] * sc[r] + rs;
    }
#pragma unroll
    for (int fn = 0; fn < 4; ++fn)
#pragma unroll
      for (int r = 0; r < 8; ++r) o[fn][r] *= sc[r];

    // P: C layout -> per-wave LDS -> A layout (same-wave DS ops in-order)
#pragma unroll
    for (int fn = 0; fn < 4; ++fn)
#pragma unroll
      for (int r = 0; r < 8; ++r)
        Ps[w][rowhalf + r][fn * 16 + fr] = (bf16_t)s[fn][r];
    v16bf pa0 = ld_afrag(&Ps[w][fr][0], kg);
    v16bf pa1 = ld_afrag(&Ps[w][fr][32], kg);

    // O += P V : B-matrix element [k=krow][n=hd] = Vs[hd][krow]
#pragma unroll
    for (int fn = 0; fn < 4; ++fn) {
      o[fn] = wmma_bf16(pa0, ld_bfrag(&Vs[fn * 16 + fr][kh]), o[fn]);
      o[fn] = wmma_bf16(pa1, ld_bfrag(&Vs[fn * 16 + fr][32 + kh]), o[fn]);
    }
    __syncthreads();
  }

  // normalize and scatter into [b, t, h*HD + c] (bf16 for the out-proj GEMM)
#pragma unroll
  for (int fn = 0; fn < 4; ++fn) {
    const int colh = hh * HD + fn * 16 + fr;
#pragma unroll
    for (int r = 0; r < 8; ++r) {
      const int trow = qt * 64 + w * 16 + rowhalf + r;
      Ctx[((size_t)(bb * SEQ + trow)) * DM + colh] = (bf16_t)(o[fn][r] / l_[r]);
    }
  }
}

// ---------------------------------------------------------------------------
// Elementwise / reduction kernels
// ---------------------------------------------------------------------------
__global__ void __launch_bounds__(256)
embed_kernel(const int* __restrict__ idx, const float* __restrict__ tok,
             const float* __restrict__ pos, float* __restrict__ x) {
  const size_t i = (size_t)blockIdx.x * 256 + threadIdx.x;
  const int bt = (int)(i >> 10);
  const int d = (int)(i & (DM - 1));
  const int t = bt & (SEQ - 1);
  x[i] = tok[(size_t)idx[bt] * DM + d] + pos[(size_t)t * DM + d];
}

__global__ void __launch_bounds__(256)
layernorm_kernel(const float* __restrict__ x, const float* __restrict__ g,
                 const float* __restrict__ b, bf16_t* __restrict__ out) {
  const int row = blockIdx.x * 8 + (threadIdx.x >> 5);  // 1 wave per row
  const int lane = threadIdx.x & 31;
  const float* xr = x + (size_t)row * DM;
  float v[32];
  float sum = 0.0f;
#pragma unroll
  for (int i = 0; i < 8; ++i) {
    const float4 f = ((const float4*)xr)[lane + i * 32];
    v[i * 4 + 0] = f.x; v[i * 4 + 1] = f.y;
    v[i * 4 + 2] = f.z; v[i * 4 + 3] = f.w;
    sum += f.x + f.y + f.z + f.w;
  }
#pragma unroll
  for (int off = 1; off < 32; off <<= 1) sum += __shfl_xor(sum, off, 32);
  const float mean = sum * (1.0f / DM);
  float var = 0.0f;
#pragma unroll
  for (int i = 0; i < 32; ++i) { const float d0 = v[i] - mean; var += d0 * d0; }
#pragma unroll
  for (int off = 1; off < 32; off <<= 1) var += __shfl_xor(var, off, 32);
  const float rinv = rsqrtf(var * (1.0f / DM) + 1e-5f);
#pragma unroll
  for (int i = 0; i < 8; ++i)
#pragma unroll
    for (int j = 0; j < 4; ++j) {
      const int d = 4 * (lane + i * 32) + j;
      out[(size_t)row * DM + d] =
          (bf16_t)((v[i * 4 + j] - mean) * rinv * g[d] + b[d]);
    }
}

// f32 -> bf16 straight copy (used for the tied lm-head: [V,D] is already [N,K])
__global__ void __launch_bounds__(256)
cvt_kernel(const float* __restrict__ in, bf16_t* __restrict__ out) {
  const size_t i = (size_t)blockIdx.x * 256 + threadIdx.x;
  out[i] = (bf16_t)in[i];
}

// f32 [K,N] -> bf16 [N,K] (weights pre-transposed for the GEMM B operand)
__global__ void __launch_bounds__(256)
cvt_transpose_kernel(const float* __restrict__ in, bf16_t* __restrict__ out,
                     int Kd, int Nd) {
  const size_t i = (size_t)blockIdx.x * 256 + threadIdx.x;
  const int k = (int)(i / (size_t)Nd);
  const int n = (int)(i % (size_t)Nd);
  out[(size_t)n * Kd + k] = (bf16_t)in[i];
}

__global__ void __launch_bounds__(256)
qkv_split_kernel(const float* __restrict__ qkv, bf16_t* __restrict__ qo,
                 bf16_t* __restrict__ ko, bf16_t* __restrict__ vo) {
  const size_t i = (size_t)blockIdx.x * 256 + threadIdx.x;
  const int bt = (int)(i >> 10);
  const int d = (int)(i & (DM - 1));
  const int bb = bt >> 11;          // / SEQ
  const int t = bt & (SEQ - 1);
  const int hh = d >> 6, e = d & 63;
  const size_t src = (size_t)bt * (3 * DM) + d;
  const size_t dst = ((size_t)(bb * NH + hh) * SEQ + t) * HD + e;
  qo[dst] = (bf16_t)qkv[src];
  ko[dst] = (bf16_t)qkv[src + DM];
  vo[dst] = (bf16_t)qkv[src + 2 * DM];
}

// ---------------------------------------------------------------------------
// Launcher
// ---------------------------------------------------------------------------
extern "C" void kernel_launch(void* const* d_in, const int* in_sizes, int n_in,
                              void* d_out, int out_size, void* d_ws,
                              size_t ws_size, hipStream_t stream) {
  const int* idx = (const int*)d_in[0];
  const float* tok = (const float*)d_in[1];
  const float* pos = (const float*)d_in[2];
  const float* qkv_w = (const float*)d_in[3];
  const float* qkv_b = (const float*)d_in[4];
  const float* out_w = (const float*)d_in[5];
  const float* out_b = (const float*)d_in[6];
  const float* ln1_s = (const float*)d_in[7];
  const float* ln1_b = (const float*)d_in[8];
  const float* ff1_w = (const float*)d_in[9];
  const float* ff1_b = (const float*)d_in[10];
  const float* ff2_w = (const float*)d_in[11];
  const float* ff2_b = (const float*)d_in[12];
  const float* ln2_s = (const float*)d_in[13];
  const float* ln2_b = (const float*)d_in[14];
  const float* lnf_s = (const float*)d_in[15];
  const float* lnf_b = (const float*)d_in[16];
  float* logits = (float*)d_out;

  char* wsp = (char*)d_ws;
  size_t off = 0;
  auto take = [&](size_t bytes) -> char* {
    char* p = wsp + off;
    off = (off + bytes + 255) & ~(size_t)255;
    return p;
  };
  float* x = (float*)take((size_t)BT * DM * 4);           // residual stream
  bf16_t* xn = (bf16_t*)take((size_t)BT * DM * 2);        // LN output
  float* qkvf = (float*)take((size_t)BT * 3 * DM * 4);    // qkv projection
  bf16_t* qb = (bf16_t*)take((size_t)BT * DM * 2);        // [B,H,T,HD]
  bf16_t* kb = (bf16_t*)take((size_t)BT * DM * 2);
  bf16_t* vb = (bf16_t*)take((size_t)BT * DM * 2);
  bf16_t* ctx = (bf16_t*)take((size_t)BT * DM * 2);       // attn context
  bf16_t* h1 = (bf16_t*)take((size_t)BT * FFD * 2);       // gelu(ff1)
  bf16_t* wbf = (bf16_t*)take((size_t)VOC * DM * 2);      // bf16 [N,K] weights

  const dim3 blk256(256), blk128(128);
  embed_kernel<<<BT * DM / 256, blk256, 0, stream>>>(idx, tok, pos, x);

  for (int l = 0; l < NL; ++l) {
    layernorm_kernel<<<BT / 8, blk256, 0, stream>>>(x, ln1_s + l * DM,
                                                    ln1_b + l * DM, xn);
    cvt_transpose_kernel<<<DM * 3 * DM / 256, blk256, 0, stream>>>(
        qkv_w + (size_t)l * DM * 3 * DM, wbf, DM, 3 * DM);
    gemm_bf16<true, false, false, false>
        <<<dim3(3 * DM / 128, BT / 128), blk256, 0, stream>>>(
            xn, wbf, qkv_b + l * 3 * DM, nullptr, qkvf, BT, 3 * DM, DM);
    qkv_split_kernel<<<BT * DM / 256, blk256, 0, stream>>>(qkvf, qb, kb, vb);
    attention_kernel<<<dim3(SEQ / 64, BATCH * NH), blk128, 0, stream>>>(
        qb, kb, vb, ctx);
    cvt_transpose_kernel<<<DM * DM / 256, blk256, 0, stream>>>(
        out_w + (size_t)l * DM * DM, wbf, DM, DM);
    gemm_bf16<true, true, false, false>
        <<<dim3(DM / 128, BT / 128), blk256, 0, stream>>>(
            ctx, wbf, out_b + l * DM, x, x, BT, DM, DM);
    layernorm_kernel<<<BT / 8, blk256, 0, stream>>>(x, ln2_s + l * DM,
                                                    ln2_b + l * DM, xn);
    cvt_transpose_kernel<<<DM * FFD / 256, blk256, 0, stream>>>(
        ff1_w + (size_t)l * DM * FFD, wbf, DM, FFD);
    gemm_bf16<true, false, true, true>
        <<<dim3(FFD / 128, BT / 128), blk256, 0, stream>>>(
            xn, wbf, ff1_b + l * FFD, nullptr, h1, BT, FFD, DM);
    cvt_transpose_kernel<<<FFD * DM / 256, blk256, 0, stream>>>(
        ff2_w + (size_t)l * FFD * DM, wbf, FFD, DM);
    gemm_bf16<true, true, false, false>
        <<<dim3(DM / 128, BT / 128), blk256, 0, stream>>>(
            h1, wbf, ff2_b + l * DM, x, x, BT, DM, FFD);
  }

  layernorm_kernel<<<BT / 8, blk256, 0, stream>>>(x, lnf_s, lnf_b, xn);
  cvt_kernel<<<(int)((size_t)VOC * DM / 256), blk256, 0, stream>>>(tok, wbf);
  gemm_bf16<false, false, false, false>
      <<<dim3(VOC / 128, BT / 128), blk256, 0, stream>>>(
          xn, wbf, nullptr, nullptr, logits, BT, VOC, DM);
}
